// PureCounter_13022340842108
// MI455X (gfx1250) — compile-verified, matrix operand order checked by reading, not run on hardware
//
#include <hip/hip_runtime.h>
#include <hip/hip_bf16.h>
#include <cstdint>

typedef __attribute__((ext_vector_type(16))) _Float16 v16h;
typedef __attribute__((ext_vector_type(8)))  float    v8f;
typedef __attribute__((ext_vector_type(4)))  int      i32x4;

#define AS1 __attribute__((address_space(1)))
#define AS3 __attribute__((address_space(3)))

// ---- gfx1250 async global->LDS path, gated on builtin availability ----
#if defined(__has_builtin)
#  if __has_builtin(__builtin_amdgcn_global_load_async_to_lds_b128) && \
      __has_builtin(__builtin_amdgcn_s_wait_asynccnt)
#    define USE_ASYNC_LDS 1
#  endif
#endif
#ifndef USE_ASYNC_LDS
#  define USE_ASYNC_LDS 0
#endif

constexpr int T_ELEMS        = 1 << 24;          // 16777216
constexpr int SEG            = 1024;             // elements per wave-segment
constexpr int NSEG           = T_ELEMS / SEG;    // 16384 segments
constexpr int WAVES_PER_BLK  = 8;
constexpr int BLOCK          = WAVES_PER_BLK * 32;   // 256 threads = 8 waves (wave32)
constexpr int NBLK           = NSEG / WAVES_PER_BLK; // 2048 blocks
constexpr int SCAN_THREADS   = 1024;
constexpr int PER_SCAN_THR   = NSEG / SCAN_THREADS;  // 16
constexpr int CHUNK          = 128;                  // elements per wave-iteration
constexpr int NCHUNK         = SEG / CHUNK;          // 8

// ---------- wave32 helpers (packed u64 scans via 32-bit shuffles) ----------
__device__ __forceinline__ unsigned long long shfl_up_u64(unsigned long long v, int d) {
  unsigned lo = __shfl_up((unsigned)v, d, 32);
  unsigned hi = __shfl_up((unsigned)(v >> 32), d, 32);
  return ((unsigned long long)hi << 32) | lo;
}
__device__ __forceinline__ unsigned long long bcast31_u64(unsigned long long v) {
  unsigned lo = __shfl((unsigned)v, 31, 32);
  unsigned hi = __shfl((unsigned)(v >> 32), 31, 32);
  return ((unsigned long long)hi << 32) | lo;
}
__device__ __forceinline__ unsigned long long wave_iscan_u64(unsigned long long v) {
  const int lane = threadIdx.x & 31;
#pragma unroll
  for (int d = 1; d < 32; d <<= 1) {
    unsigned long long u = shfl_up_u64(v, d);
    if (lane >= d) v += u;
  }
  return v;
}
// packed indicator: n0 in [31:0], n1 in [63:32]; counts <= 2^24 so no carry mixing
__device__ __forceinline__ unsigned long long ind_pack(int s) {
  return (s == 0) ? 1ull : (s == 1) ? (1ull << 32) : 0ull;
}

// ---------------- Pass 1: per-segment totals via WMMA reduction ----------------
// Indicators (0/1) packed as f16 into the 16x32 A operand; B is all-ones, so
// sum(D) = 16 * sum(A) regardless of A/B VGPR lane layouts (layout-invariant,
// exact in f16-in / f32-acc for 0/1 values).
__global__ __launch_bounds__(BLOCK) void seg_reduce_wmma(
    const int* __restrict__ in, unsigned long long* __restrict__ seg_sums) {
  const int wid  = threadIdx.x >> 5;
  const int lane = threadIdx.x & 31;
  const int seg  = blockIdx.x * WAVES_PER_BLK + wid;
  const int* p   = in + (size_t)seg * SEG;

  v16h ones;
#pragma unroll
  for (int i = 0; i < 16; ++i) ones[i] = (_Float16)1.0f;

  v8f acc0 = {};   // counts of symbol 0
  v8f acc1 = {};   // counts of symbol 1

#pragma unroll
  for (int g = 0; g < SEG / 512; ++g) {   // 512 elements per WMMA pair
    v16h a0, a1;
#pragma unroll
    for (int j = 0; j < 16; ++j) {        // coalesced: lane-consecutive loads
      int s = p[g * 512 + j * 32 + lane];
      a0[j] = (s == 0) ? (_Float16)1.0f : (_Float16)0.0f;
      a1[j] = (s == 1) ? (_Float16)1.0f : (_Float16)0.0f;
    }
    acc0 = __builtin_amdgcn_wmma_f32_16x16x32_f16(false, a0, false, ones,
                                                  (short)0, acc0, false, false);
    acc1 = __builtin_amdgcn_wmma_f32_16x16x32_f16(false, a1, false, ones,
                                                  (short)0, acc1, false, false);
  }

  float s0 = 0.f, s1 = 0.f;
#pragma unroll
  for (int i = 0; i < 8; ++i) { s0 += acc0[i]; s1 += acc1[i]; }
#pragma unroll
  for (int d = 16; d >= 1; d >>= 1) {
    s0 += __shfl_xor(s0, d, 32);
    s1 += __shfl_xor(s1, d, 32);
  }
  if (lane == 0) {
    // every D row appears 16x across columns -> divide by 16 (exact, ints <= 16384)
    unsigned long long n0 = (unsigned long long)(unsigned)(s0 * 0.0625f + 0.5f);
    unsigned long long n1 = (unsigned long long)(unsigned)(s1 * 0.0625f + 0.5f);
    seg_sums[seg] = n0 | (n1 << 32);
  }
}

// ---------------- Pass 2: exclusive scan of 16384 segment sums ----------------
__global__ __launch_bounds__(SCAN_THREADS) void scan_seg_sums(
    const unsigned long long* __restrict__ sums,
    unsigned long long* __restrict__ offs) {
  __shared__ unsigned long long wtot[SCAN_THREADS / 32];
  const int tid  = threadIdx.x;
  const int lane = tid & 31;
  const int wid  = tid >> 5;
  const int base = tid * PER_SCAN_THR;

  unsigned long long loc[PER_SCAN_THR];
  unsigned long long run = 0;
#pragma unroll
  for (int i = 0; i < PER_SCAN_THR; ++i) { loc[i] = sums[base + i]; run += loc[i]; }

  unsigned long long incl = wave_iscan_u64(run);
  if (lane == 31) wtot[wid] = incl;
  __syncthreads();
  if (wid == 0) {
    unsigned long long w  = wtot[lane];
    unsigned long long wi = wave_iscan_u64(w);
    wtot[lane] = wi - w;                 // exclusive wave offsets
  }
  __syncthreads();

  unsigned long long r = wtot[wid] + (incl - run);   // thread-exclusive prefix
#pragma unroll
  for (int i = 0; i < PER_SCAN_THR; ++i) { offs[base + i] = r; r += loc[i]; }
}

// ---------------- Pass 3: per-element scan + sigmoid emit ----------------
// Input is staged through LDS with depth-2 double-buffered async copies
// (GLOBAL_LOAD_ASYNC_TO_LDS_B128 + s_wait_asynccnt) when the toolchain exposes
// the builtins; otherwise falls back to direct B128 loads + global_prefetch_b8.
__global__ __launch_bounds__(BLOCK) void scan_emit(
    const int* __restrict__ in,
    const unsigned long long* __restrict__ seg_offs,
    const float* __restrict__ da, const float* __restrict__ db,
    const float* __restrict__ dc, float2* __restrict__ out) {
#if USE_ASYNC_LDS
  __shared__ alignas(16) int stage[WAVES_PER_BLK][2][CHUNK];   // 8 KB / block
#endif
  const int wid     = threadIdx.x >> 5;
  const int lane    = threadIdx.x & 31;
  const int seg     = blockIdx.x * WAVES_PER_BLK + wid;
  const int segbase = seg * SEG;

  double Kab[3], Kac[3];
#pragma unroll
  for (int j = 0; j < 3; ++j) {
    double a = (double)da[j];
    Kab[j] = a - (double)db[j];
    Kac[j] = a - (double)dc[j];
  }

  unsigned long long carry = seg_offs[seg];
  const int* p = in + segbase;

#if USE_ASYNC_LDS
  // prologue: stage chunk 0 (each lane copies 16B -> 512B per wave chunk)
  __builtin_amdgcn_global_load_async_to_lds_b128(
      (AS1 i32x4*)(p + lane * 4),
      (AS3 i32x4*)&stage[wid][0][lane * 4], 0, 0);
#endif

#pragma unroll
  for (int it = 0; it < NCHUNK; ++it) {    // 128 elements / wave / iter
    const int lbase = it * CHUNK + lane * 4;

#if USE_ASYNC_LDS
    if (it + 1 < NCHUNK) {
      __builtin_amdgcn_global_load_async_to_lds_b128(
          (AS1 i32x4*)(p + lbase + CHUNK),
          (AS3 i32x4*)&stage[wid][(it + 1) & 1][lane * 4], 0, 0);
      __builtin_amdgcn_s_wait_asynccnt(1);   // oldest buffer has landed (in-order)
    } else {
      __builtin_amdgcn_s_wait_asynccnt(0);
    }
    const int4 v = *(const int4*)&stage[wid][it & 1][lane * 4];  // ds_load_b128
#else
    if (it + 1 < NCHUNK)
      __builtin_prefetch(p + lbase + CHUNK, 0, 0);   // global_prefetch_b8
    const int4 v = *(const int4*)(p + lbase);        // global_load_b128
#endif

    unsigned long long s_incl[4];
    unsigned long long r = ind_pack(v.x); s_incl[0] = r;
    r += ind_pack(v.y);                   s_incl[1] = r;
    r += ind_pack(v.z);                   s_incl[2] = r;
    r += ind_pack(v.w);                   s_incl[3] = r;

    const unsigned long long wincl = wave_iscan_u64(r);
    const unsigned long long boff  = carry + (wincl - r);

#pragma unroll
    for (int i = 0; i < 4; ++i) {
      const unsigned long long cnt = boff + s_incl[i];
      const int t_idx = segbase + lbase + i;
      const double n0 = (double)(unsigned)(cnt & 0xffffffffull);
      const double n1 = (double)(unsigned)(cnt >> 32);
      const double n2 = (double)(t_idx + 1) - n0 - n1;
      const float x1 = (float)((n0 * Kab[0] + n1 * Kab[1] + n2 * Kab[2] + 0.5) * 10.0);
      const float x2 = (float)((n0 * Kac[0] + n1 * Kac[1] + n2 * Kac[2] + 0.5) * 10.0);
      const float v1 = 1.0f / (1.0f + __expf(-x1));
      const float v2 = 1.0f / (1.0f + __expf(-x2));
      const float vp = v1 * v2;
      out[t_idx] = make_float2(1.0f - vp, vp);     // contiguous float2 stores
    }
    carry += bcast31_u64(wincl);
  }
}

extern "C" void kernel_launch(void* const* d_in, const int* in_sizes, int n_in,
                              void* d_out, int out_size, void* d_ws, size_t ws_size,
                              hipStream_t stream) {
  (void)in_sizes; (void)n_in; (void)out_size; (void)ws_size;
  const int*   in = (const int*)d_in[0];
  const float* da = (const float*)d_in[1];
  const float* db = (const float*)d_in[2];
  const float* dc = (const float*)d_in[3];

  unsigned long long* seg_sums = (unsigned long long*)d_ws;        // 128 KB
  unsigned long long* seg_offs = seg_sums + NSEG;                  // 128 KB

  seg_reduce_wmma<<<NBLK, BLOCK, 0, stream>>>(in, seg_sums);
  scan_seg_sums<<<1, SCAN_THREADS, 0, stream>>>(seg_sums, seg_offs);
  scan_emit<<<NBLK, BLOCK, 0, stream>>>(in, seg_offs, da, db, dc, (float2*)d_out);
}
